// SqueezeNet_9268539425356
// MI455X (gfx1250) — 8-bit matrix operand order repaired and checked by reading, compiled, not run on hardware
//
#include <hip/hip_runtime.h>
#include <stdint.h>

typedef int v8i __attribute__((ext_vector_type(8)));
typedef int v2i __attribute__((ext_vector_type(2)));

#define EPSBN 1e-3f
#define AS_GLOBAL __attribute__((address_space(1)))
#define AS_LDS    __attribute__((address_space(3)))

__device__ __forceinline__ void async_copy_b64(const int* gsrc, int* ldst) {
#if __has_builtin(__builtin_amdgcn_global_load_async_to_lds_b64)
  __builtin_amdgcn_global_load_async_to_lds_b64(
      (AS_GLOBAL v2i*)gsrc,
      (AS_LDS    v2i*)ldst, 0, 0);
#else
  *(int2*)ldst = *(const int2*)gsrc;
#endif
}

__device__ __forceinline__ void async_wait0() {
#if __has_builtin(__builtin_amdgcn_s_wait_asynccnt)
  __builtin_amdgcn_s_wait_asynccnt(0);
#elif __has_builtin(__builtin_amdgcn_global_load_async_to_lds_b64)
  asm volatile("s_wait_asynccnt 0x0" ::: "memory");
#endif
}

__global__ void __launch_bounds__(256) stem_kernel(
    const float* __restrict__ x, const float* __restrict__ wt,
    const float* __restrict__ bias,
    const float* __restrict__ g, const float* __restrict__ b,
    const float* __restrict__ m, const float* __restrict__ v,
    int8_t* __restrict__ out)
{
  const int HH = 224, WW = 224;
  int p = blockIdx.x * blockDim.x + threadIdx.x;
  if (p >= 32 * HH * WW) return;
  int wp = p % WW;
  int t  = p / WW;
  int hp = t % HH;
  int n  = t / HH;

  float acc[32];
#pragma unroll
  for (int c = 0; c < 32; ++c) acc[c] = bias[c];

  for (int ky = 0; ky < 3; ++ky) {
    int yy = hp + ky - 1;
    if (yy < 0 || yy >= HH) continue;
    for (int kx = 0; kx < 3; ++kx) {
      int xx = wp + kx - 1;
      if (xx < 0 || xx >= WW) continue;
      const float* xp = x + (((size_t)n * HH + yy) * WW + xx) * 3;
      float x0 = xp[0], x1 = xp[1], x2 = xp[2];
      const float* wq = wt + (size_t)(ky * 3 + kx) * 96;
#pragma unroll
      for (int c = 0; c < 32; ++c)
        acc[c] = fmaf(x0, wq[c], fmaf(x1, wq[32 + c], fmaf(x2, wq[64 + c], acc[c])));
    }
  }

  int8_t* op = out + (size_t)p * 64;
#pragma unroll
  for (int c = 0; c < 32; ++c) {
    float yv = fmaxf(acc[c], 0.f);
    float z  = g[c] * (yv - m[c]) * rsqrtf(v[c] + EPSBN) + b[c];
    op[c] = (z >= 0.f) ? (int8_t)1 : (int8_t)-1;
  }
#pragma unroll
  for (int c = 32; c < 64; ++c) op[c] = 0;
}

__global__ void __launch_bounds__(256) pack_weights_kernel(
    const float* __restrict__ w, int* __restrict__ bpk,
    int taps3, int cin, int cout, int ktiles, int ntiles)
{
  int idx = blockIdx.x * blockDim.x + threadIdx.x;
  int total = ntiles * ktiles * 256;
  if (idx >= total) return;
  int j    = idx & 7;
  int lane = (idx >> 3) & 31;
  int rest = idx >> 8;
  int kt   = rest % ktiles;
  int nt   = rest / ktiles;
  int n    = nt * 16 + (lane & 15);

  unsigned word = 0;
#pragma unroll
  for (int by = 0; by < 4; ++by) {
    int klocal = ((j >> 1) << 4) + ((lane >> 4) << 3) + ((j & 1) << 2) + by;
    int tap, ch;
    if (taps3) { tap = kt; ch = klocal; }
    else       { tap = 0;  ch = kt * 64 + klocal; }
    int val = 0;
    if (ch < cin && n < cout) {
      float wv = w[((size_t)tap * cin + ch) * cout + n];
      val = (wv >= 0.f) ? 1 : -1;
    }
    word |= ((unsigned)(val & 0xFF)) << (8 * by);
  }
  bpk[idx] = (int)word;
}

__global__ void __launch_bounds__(32) bconv_wmma_kernel(
    const int8_t* __restrict__ act, const int* __restrict__ bpk,
    const float* __restrict__ bn_g, const float* __restrict__ bn_b,
    const float* __restrict__ bn_m, const float* __restrict__ bn_v,
    int H, int W, int Cpad, int ktiles, int taps3, int cout, int ntiles,
    int8_t* __restrict__ out_bin, int outCpad,
    float* __restrict__ out_f, int outC, int outChOff, int write_float)
{
  __shared__ int ldsB[2 * 9 * 256];

  const int lane  = threadIdx.x;
  const int rloc  = lane & 15;
  const int koff8 = (lane >> 4) << 3;
  const long row0 = (long)blockIdx.x * 64;
  const int  nt0  = blockIdx.y * 2;
  const int  nts  = (nt0 + 1 < ntiles) ? 2 : 1;

  {
    const int totInts = nts * ktiles * 256;
    const int* gB = bpk + (size_t)nt0 * ktiles * 256;
    for (int o = lane * 2; o < totInts; o += 64)
      async_copy_b64(gB + o, ldsB + o);
    async_wait0();
  }

  int ph[4], pw[4];
  size_t rbase[4];
#pragma unroll
  for (int s = 0; s < 4; ++s) {
    int p = (int)row0 + s * 16 + rloc;
    if (taps3) {
      pw[s] = p % W;
      int tq = p / W;
      ph[s] = tq % H;
      int n = tq / H;
      rbase[s] = (size_t)n * H * W;
    } else {
      rbase[s] = (size_t)p * Cpad + koff8;
      ph[s] = 0; pw[s] = 0;
    }
  }

  const v8i vzero = {0, 0, 0, 0, 0, 0, 0, 0};
  v8i acc[8];
#pragma unroll
  for (int s = 0; s < 8; ++s) acc[s] = vzero;

  for (int kt = 0; kt < ktiles; ++kt) {
    v8i bb0 = *(const v8i*)(ldsB + (size_t)kt * 256 + lane * 8);
    v8i bb1 = bb0;
    if (nts == 2)
      bb1 = *(const v8i*)(ldsB + (size_t)(ktiles + kt) * 256 + lane * 8);

    int dy = 0, dx = 0;
    if (taps3) { dy = kt / 3 - 1; dx = kt % 3 - 1; }

#pragma unroll
    for (int s = 0; s < 4; ++s) {
      v8i aa = vzero;
      if (taps3) {
        int hh = ph[s] + dy, ww = pw[s] + dx;
        if (hh >= 0 && hh < H && ww >= 0 && ww < W) {
          const int8_t* arow =
              act + (rbase[s] + (size_t)hh * W + ww) * Cpad + koff8;
          uint2 d0 = *(const uint2*)(arow);
          uint2 d1 = *(const uint2*)(arow + 16);
          uint2 d2 = *(const uint2*)(arow + 32);
          uint2 d3 = *(const uint2*)(arow + 48);
          aa[0] = (int)d0.x; aa[1] = (int)d0.y; aa[2] = (int)d1.x; aa[3] = (int)d1.y;
          aa[4] = (int)d2.x; aa[5] = (int)d2.y; aa[6] = (int)d3.x; aa[7] = (int)d3.y;
        }
      } else {
        const int8_t* arow = act + rbase[s] + (size_t)kt * 64;
        uint2 d0 = *(const uint2*)(arow);
        uint2 d1 = *(const uint2*)(arow + 16);
        uint2 d2 = *(const uint2*)(arow + 32);
        uint2 d3 = *(const uint2*)(arow + 48);
        aa[0] = (int)d0.x; aa[1] = (int)d0.y; aa[2] = (int)d1.x; aa[3] = (int)d1.y;
        aa[4] = (int)d2.x; aa[5] = (int)d2.y; aa[6] = (int)d3.x; aa[7] = (int)d3.y;
      }
      acc[s] = __builtin_amdgcn_wmma_i32_16x16x64_iu8(
          true, aa, true, bb0, acc[s], false, false);
      if (nts == 2)
        acc[4 + s] = __builtin_amdgcn_wmma_i32_16x16x64_iu8(
            true, aa, true, bb1, acc[4 + s], false, false);
    }
  }

  const int mbase = (lane < 16) ? 0 : 8;
  for (int t = 0; t < nts; ++t) {
    int c = (nt0 + t) * 16 + rloc;
    float gg = 0.f, bb2 = 0.f, mm = 0.f, iv = 0.f;
    if (c < cout) {
      gg = bn_g[c]; bb2 = bn_b[c]; mm = bn_m[c];
      iv = rsqrtf(bn_v[c] + EPSBN);
    }
#pragma unroll
    for (int s = 0; s < 4; ++s) {
#pragma unroll
      for (int r = 0; r < 8; ++r) {
        long p = row0 + s * 16 + mbase + r;
        float xv = fmaxf((float)acc[t * 4 + s][r], 0.f);
        float yv = gg * (xv - mm) * iv + bb2;
        if (write_float) {
          if (c < cout) out_f[(size_t)p * outC + outChOff + c] = yv;
        } else {
          out_bin[(size_t)p * outCpad + c] =
              (c < cout) ? ((yv >= 0.f) ? (int8_t)1 : (int8_t)-1) : (int8_t)0;
        }
      }
    }
  }
}

__global__ void __launch_bounds__(256) pool_sign_kernel(
    const float* __restrict__ in, int H, int W, int C,
    int8_t* __restrict__ out, int Cpad, int H2, int W2)
{
  size_t total = (size_t)32 * H2 * W2 * Cpad;
  size_t idx = (size_t)blockIdx.x * blockDim.x + threadIdx.x;
  if (idx >= total) return;
  int c  = (int)(idx % Cpad);
  size_t t = idx / Cpad;
  int w2 = (int)(t % W2); t /= W2;
  int h2 = (int)(t % H2);
  int n  = (int)(t / H2);

  int8_t r = 0;
  if (c < C) {
    const float* p0 = in + (((size_t)n * H + 2 * h2) * W + 2 * w2) * C + c;
    size_t rs = (size_t)W * C;
    float mx = fmaxf(fmaxf(p0[0], p0[C]), fmaxf(p0[rs], p0[rs + C]));
    r = (mx >= 0.f) ? (int8_t)1 : (int8_t)-1;
  }
  out[idx] = r;
}

__global__ void __launch_bounds__(256) head_kernel(
    const float* __restrict__ F, const float* __restrict__ fcw,
    float* __restrict__ out)
{
  int n = blockIdx.x;
  int t = threadIdx.x;
  __shared__ float y[48];
  __shared__ float red[256];

  if (t < 48) {
    float s = 0.f;
    const float* fp = F + (size_t)n * 196 * 48 + t;
    for (int i = 0; i < 196; ++i) s += fp[(size_t)i * 48];
    y[t] = s * (1.f / 196.f);
  }
  __syncthreads();

  float lg[4];
  float lmax = -3.0e38f;
#pragma unroll
  for (int q = 0; q < 4; ++q) {
    int o = t * 4 + q;
    float d = -3.0e38f;
    if (o < 1000) {
      d = 0.f;
#pragma unroll
      for (int c = 0; c < 48; ++c) d = fmaf(y[c], fcw[(size_t)c * 1000 + o], d);
    }
    lg[q] = d;
    lmax = fmaxf(lmax, d);
  }

  red[t] = lmax;
  __syncthreads();
  for (int off = 128; off > 0; off >>= 1) {
    if (t < off) red[t] = fmaxf(red[t], red[t + off]);
    __syncthreads();
  }
  float gmax = red[0];
  __syncthreads();

  float lsum = 0.f;
#pragma unroll
  for (int q = 0; q < 4; ++q) {
    int o = t * 4 + q;
    if (o < 1000) { lg[q] = __expf(lg[q] - gmax); lsum += lg[q]; }
    else lg[q] = 0.f;
  }
  red[t] = lsum;
  __syncthreads();
  for (int off = 128; off > 0; off >>= 1) {
    if (t < off) red[t] += red[t + off];
    __syncthreads();
  }
  float inv = 1.f / red[0];
#pragma unroll
  for (int q = 0; q < 4; ++q) {
    int o = t * 4 + q;
    if (o < 1000) out[(size_t)n * 1000 + o] = lg[q] * inv;
  }
}

extern "C" void kernel_launch(void* const* d_in, const int* in_sizes, int n_in,
                              void* d_out, int out_size, void* d_ws, size_t ws_size,
                              hipStream_t stream)
{
  (void)in_sizes; (void)n_in; (void)out_size; (void)ws_size;

  int ii = 0;
  const float* X    = (const float*)d_in[ii++];
  const float* c1w  = (const float*)d_in[ii++];
  const float* c1b  = (const float*)d_in[ii++];
  const float* b1g  = (const float*)d_in[ii++];
  const float* b1bb = (const float*)d_in[ii++];
  const float* b1m  = (const float*)d_in[ii++];
  const float* b1v  = (const float*)d_in[ii++];

  const float *fw1[5], *fw2[5], *fw3[5];
  const float *g1[5], *q1[5], *m1[5], *v1[5];
  const float *g2[5], *q2[5], *m2[5], *v2[5];
  const float *g3[5], *q3[5], *m3[5], *v3[5];
  for (int i = 0; i < 5; ++i) {
    fw1[i] = (const float*)d_in[ii++];
    g1[i] = (const float*)d_in[ii++]; q1[i] = (const float*)d_in[ii++];
    m1[i] = (const float*)d_in[ii++]; v1[i] = (const float*)d_in[ii++];
    fw2[i] = (const float*)d_in[ii++];
    g2[i] = (const float*)d_in[ii++]; q2[i] = (const float*)d_in[ii++];
    m2[i] = (const float*)d_in[ii++]; v2[i] = (const float*)d_in[ii++];
    fw3[i] = (const float*)d_in[ii++];
    g3[i] = (const float*)d_in[ii++]; q3[i] = (const float*)d_in[ii++];
    m3[i] = (const float*)d_in[ii++]; v3[i] = (const float*)d_in[ii++];
  }
  const float* fcw = (const float*)d_in[ii++];

  char* ws = (char*)d_ws;
  int8_t* A  = (int8_t*)ws;
  int8_t* S  = (int8_t*)(ws + (size_t)128 * 1024 * 1024);
  float*  F  = (float*) (ws + (size_t)256 * 1024 * 1024);
  int*    WB = (int*)   (ws + (size_t)576 * 1024 * 1024);

  {
    int P = 32 * 224 * 224;
    stem_kernel<<<dim3((P + 255) / 256), dim3(256), 0, stream>>>(
        X, c1w, c1b, b1g, b1bb, b1m, b1v, A);
  }

  static const int FS[5] = {24, 48, 64, 48, 24};
  static const int FE[5] = {24, 48, 64, 48, 24};

  int H = 224, W = 224, cin = 32, cinPad = 64;
  for (int i = 0; i < 5; ++i) {
    int s = FS[i], e = FE[i];
    int P = 32 * H * W;
    int mblk = P / 64;

    {
      int kt = cinPad / 64, ntl = 4;
      int tot = ntl * kt * 256;
      pack_weights_kernel<<<dim3((tot + 255) / 256), dim3(256), 0, stream>>>(
          fw1[i], WB, 0, cin, s, kt, ntl);
      bconv_wmma_kernel<<<dim3(mblk, (ntl + 1) / 2), dim3(32), 0, stream>>>(
          A, WB, g1[i], q1[i], m1[i], v1[i],
          H, W, cinPad, kt, 0, s, ntl,
          S, 64, nullptr, 0, 0, 0);
    }

    int ntE = (e + 15) / 16;

    {
      int tot = ntE * 1 * 256;
      pack_weights_kernel<<<dim3((tot + 255) / 256), dim3(256), 0, stream>>>(
          fw2[i], WB, 0, s, e, 1, ntE);
      bconv_wmma_kernel<<<dim3(mblk, (ntE + 1) / 2), dim3(32), 0, stream>>>(
          S, WB, g2[i], q2[i], m2[i], v2[i],
          H, W, 64, 1, 0, e, ntE,
          nullptr, 0, F, 2 * e, 0, 1);
    }

    {
      int tot = ntE * 9 * 256;
      pack_weights_kernel<<<dim3((tot + 255) / 256), dim3(256), 0, stream>>>(
          fw3[i], WB, 1, s, e, 9, ntE);
      bconv_wmma_kernel<<<dim3(mblk, (ntE + 1) / 2), dim3(32), 0, stream>>>(
          S, WB, g3[i], q3[i], m3[i], v3[i],
          H, W, 64, 9, 1, e, ntE,
          nullptr, 0, F, 2 * e, e, 1);
    }

    int C = 2 * e;
    if (i < 4) {
      int H2 = H / 2, W2 = W / 2;
      int Cp2 = ((C + 63) / 64) * 64;
      size_t tot = (size_t)32 * H2 * W2 * Cp2;
      pool_sign_kernel<<<dim3((unsigned)((tot + 255) / 256)), dim3(256), 0, stream>>>(
          F, H, W, C, A, Cp2, H2, W2);
      H = H2; W = W2; cin = C; cinPad = Cp2;
    }
  }

  head_kernel<<<dim3(32), dim3(256), 0, stream>>>(F, fcw, (float*)d_out);
}
